// Net_27943057228397
// MI455X (gfx1250) — matrix operand order repaired and checked by reading, not compiled, not run on hardware
//
#include <hip/hip_runtime.h>

typedef __attribute__((ext_vector_type(16))) _Float16 v16h;
typedef __attribute__((ext_vector_type(8)))  float    v8f;

#define BATCH 4096
#define HLEN  200
#define XDIM  384
#define N1OUT 200
#define N1PAD 208   // 13 * 16
#define K2PAD 224   // 7 * 32
#define N2OUT 80
#define PFD   8     // gather prefetch distance

// ---------------------------------------------------------------------------
// 1) cross-row cumulative AND of (history[b,0] != 0)  (reference "row_active")
// ---------------------------------------------------------------------------
__global__ void rowact_kernel(const int* __restrict__ history, int* __restrict__ rowact) {
    if (blockIdx.x == 0 && threadIdx.x == 0) {
        int act = 1;
        for (int b = 0; b < BATCH; ++b) {
            if (history[(size_t)b * HLEN] == 0) act = 0;
            rowact[b] = act;
        }
    }
}

// ---------------------------------------------------------------------------
// 2) embedding gather + masked-prefix mean pooling + concat -> x[B,384]
//    one block per batch row, 128 threads (one per output dim of pooled part).
//    global_prefetch of upcoming embedding rows hides L2 gather latency.
// ---------------------------------------------------------------------------
__global__ void embed_pool_kernel(const int* __restrict__ user,
                                  const int* __restrict__ item,
                                  const int* __restrict__ history,
                                  const int* __restrict__ cate_list,
                                  const int* __restrict__ rowact,
                                  const float* __restrict__ uW,
                                  const float* __restrict__ iW,
                                  const float* __restrict__ cW,
                                  float* __restrict__ x) {
    const int b = blockIdx.x;
    const int d = threadIdx.x;              // 0..127
    __shared__ int s_p;
    if (d == 0) s_p = HLEN;
    __syncthreads();
    const int* hrow = history + (size_t)b * HLEN;
    // first zero at t>=1 bounds the valid prefix (t=0 is always in prefix)
    for (int t = 1 + d; t < HLEN; t += 128)
        if (hrow[t] == 0) atomicMin(&s_p, t);
    __syncthreads();
    const int act = rowact[b];
    const int T   = s_p;                    // valid count when active (>=1)

    float acc = 0.0f;
    if (act) {
        if (d < 64) {
            for (int t = 0; t < T; ++t) {
                if (t + PFD < T)
                    __builtin_prefetch(&iW[(size_t)hrow[t + PFD] * 64 + d], 0, 1);
                acc += iW[(size_t)hrow[t] * 64 + d];
            }
        } else {
            const int dd = d - 64;
            for (int t = 0; t < T; ++t) {
                if (t + PFD < T)
                    __builtin_prefetch(&cate_list[hrow[t + PFD]], 0, 1);
                acc += cW[(size_t)cate_list[hrow[t]] * 64 + dd];
            }
        }
        acc /= (float)T;
    }
    float* xr = x + (size_t)b * XDIM;
    xr[256 + d] = acc;
    xr[d] = uW[(size_t)user[b] * 128 + d];
    if (d < 64) xr[128 + d] = iW[(size_t)item[b] * 64 + d];
    else        xr[192 + (d - 64)] = cW[(size_t)cate_list[item[b]] * 64 + (d - 64)];
}

// ---------------------------------------------------------------------------
// 3) BatchNorm batch statistics (biased var), folded into scale/shift
// ---------------------------------------------------------------------------
__global__ void bn_stats_kernel(const float* __restrict__ x,
                                const float* __restrict__ gamma,
                                const float* __restrict__ beta,
                                float* __restrict__ scale,
                                float* __restrict__ shift) {
    const int c = blockIdx.x;               // 384 blocks
    __shared__ float ssum[256];
    __shared__ float ssq[256];
    float s = 0.0f, q = 0.0f;
    for (int r = threadIdx.x; r < BATCH; r += 256) {
        float v = x[(size_t)r * XDIM + c];
        s += v; q += v * v;
    }
    ssum[threadIdx.x] = s; ssq[threadIdx.x] = q;
    __syncthreads();
    for (int st = 128; st > 0; st >>= 1) {
        if (threadIdx.x < st) {
            ssum[threadIdx.x] += ssum[threadIdx.x + st];
            ssq[threadIdx.x]  += ssq[threadIdx.x + st];
        }
        __syncthreads();
    }
    if (threadIdx.x == 0) {
        float mean = ssum[0] * (1.0f / BATCH);
        float var  = ssq[0] * (1.0f / BATCH) - mean * mean;
        float sc   = gamma[c] * rsqrtf(var + 1e-5f);
        scale[c] = sc;
        shift[c] = beta[c] - mean * sc;
    }
}

// ---------------------------------------------------------------------------
// 4) apply BN, convert to f16 for WMMA input
// ---------------------------------------------------------------------------
__global__ void bn_apply_kernel(const float* __restrict__ x,
                                const float* __restrict__ scale,
                                const float* __restrict__ shift,
                                _Float16* __restrict__ xnh) {
    const int i = blockIdx.x * 256 + threadIdx.x;    // grid covers 4096*384
    const int c = i % XDIM;
    xnh[i] = (_Float16)(x[i] * scale[c] + shift[c]);
}

// ---------------------------------------------------------------------------
// 5) weight conversion to f16 with zero padding; zero-fill y1 pad region
// ---------------------------------------------------------------------------
__global__ void prep_kernel(const float* __restrict__ W1, const float* __restrict__ W2,
                            _Float16* __restrict__ w1h, _Float16* __restrict__ w2h,
                            _Float16* __restrict__ y1h) {
    const int NW1 = N1PAD * XDIM;            // 208*384
    const int NW2 = N2OUT * K2PAD;           // 80*224
    const int NY1 = BATCH * K2PAD;           // 4096*224
    const int total = NW1 + NW2 + NY1;
    for (int i = blockIdx.x * blockDim.x + threadIdx.x; i < total;
         i += gridDim.x * blockDim.x) {
        if (i < NW1) {
            int r = i / XDIM, c = i % XDIM;
            w1h[i] = (r < N1OUT) ? (_Float16)W1[(size_t)r * XDIM + c] : (_Float16)0.0f;
        } else if (i < NW1 + NW2) {
            int j = i - NW1;
            int r = j / K2PAD, c = j % K2PAD;
            w2h[j] = (c < N1OUT) ? (_Float16)W2[(size_t)r * N1OUT + c] : (_Float16)0.0f;
        } else {
            y1h[i - NW1 - NW2] = (_Float16)0.0f;
        }
    }
}

// ---------------------------------------------------------------------------
// 6) layer 1 GEMM via WMMA: y1 = prelu(xn @ W1^T + b1), M=4096 N=208 K=384
//    one wave per 16x16 output tile, 12 k-steps of v_wmma_f32_16x16x32_f16
// ---------------------------------------------------------------------------
__global__ void gemm1_wmma_kernel(const _Float16* __restrict__ xnh,
                                  const _Float16* __restrict__ w1h,
                                  const float* __restrict__ b1,
                                  const float* __restrict__ a1p,
                                  _Float16* __restrict__ y1h) {
    const int wave = threadIdx.x >> 5;
    const int lane = threadIdx.x & 31;
    const int tile = blockIdx.x * 8 + wave;     // 416*8 = 3328 = 256*13 exactly
    const int mt = tile / 13;
    const int nt = tile - mt * 13;
    const int hi = lane >> 4;
    const int ln = lane & 15;
    const _Float16* arow = xnh + (size_t)(mt * 16 + ln) * XDIM;
    const _Float16* brow = w1h + (size_t)(nt * 16 + ln) * XDIM;

    v8f c = {};
    for (int ks = 0; ks < 12; ++ks) {
        const int k0 = ks * 32;
        v16h a, b;
#pragma unroll
        for (int j = 0; j < 8; ++j) {
            // A 16x32 f16 layout: lanes 0-15 K{0-7,16-23}, lanes 16-31 K{8-15,24-31}
            const int ka = k0 + 2 * j + ((j >= 4) ? 8 : 0) + (hi ? 8 : 0);
            a[2 * j]     = arow[ka];
            a[2 * j + 1] = arow[ka + 1];
            // B 32x16 f16 layout: lanes 0-15 K{0-7,16-23}, lanes 16-31 K{8-15,24-31}
            const int kb = k0 + 2 * j + ((j >= 4) ? 8 : 0) + (hi ? 8 : 0);
            b[2 * j]     = brow[kb];
            b[2 * j + 1] = brow[kb + 1];
        }
        c = __builtin_amdgcn_wmma_f32_16x16x32_f16(false, a, false, b,
                                                   (short)0, c, false, false);
    }
    const int n = nt * 16 + ln;
    if (n < N1OUT) {
        const float bias = b1[n];
        const float al   = a1p[0];
#pragma unroll
        for (int r = 0; r < 8; ++r) {
            const int m = mt * 16 + hi * 8 + r;   // C/D: vgpr r -> M = r + 8*hi
            float v = c[r] + bias;
            v = (v >= 0.0f) ? v : al * v;
            y1h[(size_t)m * K2PAD + n] = (_Float16)v;
        }
    }
}

// ---------------------------------------------------------------------------
// 7) layer 2 GEMM via WMMA: y2 = prelu(y1 @ W2^T + b2), M=4096 N=80 K=224
// ---------------------------------------------------------------------------
__global__ void gemm2_wmma_kernel(const _Float16* __restrict__ y1h,
                                  const _Float16* __restrict__ w2h,
                                  const float* __restrict__ b2,
                                  const float* __restrict__ a2p,
                                  float* __restrict__ y2) {
    const int wave = threadIdx.x >> 5;
    const int lane = threadIdx.x & 31;
    const int tile = blockIdx.x * 8 + wave;     // 160*8 = 1280 = 256*5 exactly
    const int mt = tile / 5;
    const int nt = tile - mt * 5;
    const int hi = lane >> 4;
    const int ln = lane & 15;
    const _Float16* arow = y1h + (size_t)(mt * 16 + ln) * K2PAD;
    const _Float16* brow = w2h + (size_t)(nt * 16 + ln) * K2PAD;

    v8f c = {};
    for (int ks = 0; ks < 7; ++ks) {
        const int k0 = ks * 32;
        v16h a, b;
#pragma unroll
        for (int j = 0; j < 8; ++j) {
            const int ka = k0 + 2 * j + ((j >= 4) ? 8 : 0) + (hi ? 8 : 0);
            a[2 * j]     = arow[ka];
            a[2 * j + 1] = arow[ka + 1];
            const int kb = k0 + 2 * j + ((j >= 4) ? 8 : 0) + (hi ? 8 : 0);
            b[2 * j]     = brow[kb];
            b[2 * j + 1] = brow[kb + 1];
        }
        c = __builtin_amdgcn_wmma_f32_16x16x32_f16(false, a, false, b,
                                                   (short)0, c, false, false);
    }
    const int n    = nt * 16 + ln;              // < 80 always
    const float bias = b2[n];
    const float al   = a2p[0];
#pragma unroll
    for (int r = 0; r < 8; ++r) {
        const int m = mt * 16 + hi * 8 + r;
        float v = c[r] + bias;
        v = (v >= 0.0f) ? v : al * v;
        y2[(size_t)m * N2OUT + n] = v;
    }
}

// ---------------------------------------------------------------------------
// 8) final 2-wide layer + softmax, one thread per row
// ---------------------------------------------------------------------------
__global__ void final_kernel(const float* __restrict__ y2,
                             const float* __restrict__ W3,
                             const float* __restrict__ b3,
                             float* __restrict__ out) {
    const int b = blockIdx.x * 128 + threadIdx.x;   // 32*128 = 4096
    const float* yr = y2 + (size_t)b * N2OUT;
    float l0 = b3[0], l1 = b3[1];
    for (int k = 0; k < N2OUT; ++k) {
        const float v = yr[k];
        l0 += W3[k] * v;
        l1 += W3[N2OUT + k] * v;
    }
    const float mx = fmaxf(l0, l1);
    const float e0 = __expf(l0 - mx);
    const float e1 = __expf(l1 - mx);
    const float inv = 1.0f / (e0 + e1);
    out[(size_t)b * 2]     = e0 * inv;
    out[(size_t)b * 2 + 1] = e1 * inv;
}

// ---------------------------------------------------------------------------
extern "C" void kernel_launch(void* const* d_in, const int* in_sizes, int n_in,
                              void* d_out, int out_size, void* d_ws, size_t ws_size,
                              hipStream_t stream) {
    (void)in_sizes; (void)n_in; (void)out_size; (void)ws_size;
    const int*   user      = (const int*)  d_in[0];
    const int*   item      = (const int*)  d_in[1];
    const int*   history   = (const int*)  d_in[2];
    // d_in[3] = length (unused; mask derived from zero padding, per reference)
    const int*   cate_list = (const int*)  d_in[4];
    const float* uW        = (const float*)d_in[5];
    const float* iW        = (const float*)d_in[6];
    const float* cW        = (const float*)d_in[7];
    const float* gamma     = (const float*)d_in[8];
    const float* beta      = (const float*)d_in[9];
    const float* W1        = (const float*)d_in[10];
    const float* b1        = (const float*)d_in[11];
    const float* a1        = (const float*)d_in[12];
    const float* W2        = (const float*)d_in[13];
    const float* b2        = (const float*)d_in[14];
    const float* a2        = (const float*)d_in[15];
    const float* W3        = (const float*)d_in[16];
    const float* b3        = (const float*)d_in[17];
    float* out = (float*)d_out;

    // workspace carve-up (256B aligned)
    char* ws = (char*)d_ws;
    size_t off = 0;
    auto carve = [&](size_t bytes) {
        char* p = ws + off;
        off = (off + bytes + 255) & ~(size_t)255;
        return p;
    };
    int*      rowact = (int*)     carve((size_t)BATCH * 4);
    float*    x      = (float*)   carve((size_t)BATCH * XDIM * 4);
    float*    scale  = (float*)   carve((size_t)XDIM * 4);
    float*    shift  = (float*)   carve((size_t)XDIM * 4);
    _Float16* xnh    = (_Float16*)carve((size_t)BATCH * XDIM * 2);
    _Float16* w1h    = (_Float16*)carve((size_t)N1PAD * XDIM * 2);
    _Float16* w2h    = (_Float16*)carve((size_t)N2OUT * K2PAD * 2);
    _Float16* y1h    = (_Float16*)carve((size_t)BATCH * K2PAD * 2);
    float*    y2     = (float*)   carve((size_t)BATCH * N2OUT * 4);

    rowact_kernel<<<1, 32, 0, stream>>>(history, rowact);
    embed_pool_kernel<<<BATCH, 128, 0, stream>>>(user, item, history, cate_list,
                                                 rowact, uW, iW, cW, x);
    bn_stats_kernel<<<XDIM, 256, 0, stream>>>(x, gamma, beta, scale, shift);
    bn_apply_kernel<<<(BATCH * XDIM) / 256, 256, 0, stream>>>(x, scale, shift, xnh);
    prep_kernel<<<1024, 256, 0, stream>>>(W1, W2, w1h, w2h, y1h);
    gemm1_wmma_kernel<<<(256 * 13) / 8, 256, 0, stream>>>(xnh, w1h, b1, a1, y1h);
    gemm2_wmma_kernel<<<(256 * 5) / 8, 256, 0, stream>>>(y1h, w2h, b2, a2, y2);
    final_kernel<<<BATCH / 128, 128, 0, stream>>>(y2, W3, b3, out);
}
